// DynamicGRU_14259291423220
// MI455X (gfx1250) — matrix operand order repaired and checked by reading, not compiled, not run on hardware
//
#include <hip/hip_runtime.h>
#include <hip/hip_bf16.h>

// ---------------------------------------------------------------------------
// DynamicGRU for MI455X (gfx1250) — persistent bf16-WMMA kernel, v3.
//   B=512, T=200, D=256, H=256, K = D+H = 512
// Each workgroup owns 16 batch rows and runs the entire time loop internally
// (the GRU recurrence is independent per batch row): 2 launches total, h state
// lives in LDS, weights stream from L2 every step.
// v3 fixes vs v2:
//   - asm memory clobber per time-step stops LICM from hoisting the (loop-
//     invariant) weight fragment loads, which had exploded VGPR usage past 256
//     and spilled to scratch inside the WMMA loop.
//   - LDS row padding (+8 bf16) on the A-fragment arrays: row stride becomes
//     260 DWORDs = 4 (mod 64 banks), so the 16 lanes of each half-wave hit
//     disjoint banks instead of a 16-way conflict on every ds_load_b128.
// Workspace: [0,512KB) WgT bf16 [512][512]; [512KB,768KB) WcT bf16 [256][512].
// ---------------------------------------------------------------------------

#define BB   512
#define TT   200
#define DD   256
#define HH   256
#define KK   512   // D + H
#define PAD  8     // bf16 elements of row padding (16B, keeps 16B alignment)

typedef __attribute__((ext_vector_type(16))) __bf16 v16bf;
typedef __attribute__((ext_vector_type(8)))  __bf16 v8bf;
typedef __attribute__((ext_vector_type(8)))  float  v8f;

static __device__ __forceinline__ v16bf cat8(v8bf lo, v8bf hi) {
  return __builtin_shufflevector(lo, hi, 0,1,2,3,4,5,6,7,8,9,10,11,12,13,14,15);
}

static __device__ __forceinline__ float sigmoidf_(float x) {
  return 1.0f / (1.0f + __expf(-x));
}

// ---------------------------------------------------------------------------
// One-shot transpose + f32->bf16 convert of both weight matrices.
// ---------------------------------------------------------------------------
__global__ void gru_prep(const float* __restrict__ gk, const float* __restrict__ ck,
                         __bf16* __restrict__ WgT, __bf16* __restrict__ WcT) {
  int i = blockIdx.x * 256 + threadIdx.x;
  if (i < KK * (2 * HH)) {
    int n = i >> 9;
    int k = i & (KK - 1);
    WgT[i] = (__bf16)gk[k * (2 * HH) + n];
  } else if (i < KK * (2 * HH) + HH * KK) {
    int j = i - KK * (2 * HH);
    int n = j >> 9;
    int k = j & (KK - 1);
    WcT[j] = (__bf16)ck[k * HH + n];
  }
}

// ---------------------------------------------------------------------------
// Persistent GRU. Grid: 32 WGs (16 batch rows each) x 256 threads (8 wave32s).
// Wave w owns gate n-tiles {w, w+8, w+16, w+24} (tiles 0..15 = r, 16..31 = u)
// and candidate n-tiles {w, w+8}.
// ---------------------------------------------------------------------------
__global__ void __launch_bounds__(256)
gru_persistent(const float* __restrict__ x,       // [B, T, D]
               const int*   __restrict__ seqlen,  // [B, 1]
               const float* __restrict__ gb,      // [2H]
               const float* __restrict__ cb,      // [H]
               const __bf16* __restrict__ WgT,    // [2H][K]
               const __bf16* __restrict__ WcT,    // [H][K]
               float* __restrict__ out) {         // [B, T, H]
  __shared__ __attribute__((aligned(16))) __bf16 sXH[16][KK + PAD];  // concat(x_t,h)
  __shared__ __attribute__((aligned(16))) __bf16 sRH[16][HH + PAD];  // r*h
  __shared__ __attribute__((aligned(16))) float  sU [16][HH];        // update gate
  __shared__ __attribute__((aligned(16))) float  sH [16][HH];        // h state
  __shared__ int sLen[16];

  const int tid  = threadIdx.x;
  const int b0   = blockIdx.x * 16;   // first batch row of this WG
  const int wave = tid >> 5;
  const int lane = tid & 31;
  const int n16  = lane & 15;         // tile column / A-row index
  const int g    = lane >> 4;         // lane half-group

  // ---- init: h = 0, cache sequence lengths ----
  {
    const int r = tid >> 4, c0 = (tid & 15) * 16;
#pragma unroll
    for (int j = 0; j < 16; ++j) sH[r][c0 + j] = 0.0f;
    if (tid < 16) sLen[tid] = seqlen[b0 + tid];
  }

  // ---- per-wave constants (registers; survive the clobber) ----
  float gbias[4];
  const __bf16* gwrow[4];
#pragma unroll
  for (int ti = 0; ti < 4; ++ti) {
    const int n = (wave + ti * 8) * 16 + n16;
    gbias[ti] = gb[n];
    gwrow[ti] = WgT + (size_t)n * KK;
  }
  float cbias[2];
  const __bf16* cwrow[2];
#pragma unroll
  for (int ti = 0; ti < 2; ++ti) {
    const int n = (wave + ti * 8) * 16 + n16;
    cbias[ti] = cb[n];
    cwrow[ti] = WcT + (size_t)n * KK;
  }

  // staging geometry (32 contiguous elems per thread; col stays in one half)
  const int srow = (tid * 32) >> 9;
  const int scol = (tid * 32) & (KK - 1);

  __syncthreads();

#pragma clang loop unroll(disable)
  for (int t = 0; t < TT; ++t) {
    // Stop LICM from hoisting weight loads across the time loop (round-2 bug:
    // hoisting spilled >256 VGPRs of weight fragments to scratch).
    asm volatile("" ::: "memory");

    // ---- stage concat(x_t, h_t) -> LDS bf16 ----
    if (scol < DD) {
      const float* src = x + (((size_t)(b0 + srow) * TT) + t) * DD + scol;
#pragma unroll
      for (int j = 0; j < 32; ++j) sXH[srow][scol + j] = (__bf16)src[j];
    } else {
#pragma unroll
      for (int j = 0; j < 32; ++j) sXH[srow][scol + j] = (__bf16)sH[srow][scol - DD + j];
    }
    __syncthreads();

    // ---- gate GEMM: [16,512] x [512,512] + bias ----
#pragma unroll
    for (int ti = 0; ti < 4; ++ti) {
      const int ntile = wave + ti * 8;        // 0..31
      const float bias = gbias[ti];
      v8f acc = {bias, bias, bias, bias, bias, bias, bias, bias};
      const __bf16* wrow = gwrow[ti];
#pragma unroll
      for (int kt = 0; kt < KK / 32; ++kt) {
        const int ka = kt * 32 + g * 8;
        v8bf a0 = *(const v8bf*)&sXH[n16][ka];
        v8bf a1 = *(const v8bf*)&sXH[n16][ka + 16];
        const int kb = kt * 32 + g * 8;
        v8bf b0 = *(const v8bf*)(wrow + kb);
        v8bf b1 = *(const v8bf*)(wrow + kb + 16);
        acc = __builtin_amdgcn_wmma_f32_16x16x32_bf16(
            false, cat8(a0, a1), false, cat8(b0, b1), (short)0, acc, false, false);
      }
      if (ntile < 16) {                        // r gate -> r*h bf16 staging
        const int col = ntile * 16 + n16;
#pragma unroll
        for (int vr = 0; vr < 8; ++vr) {
          const int m = vr + 8 * g;
          float s = sigmoidf_(acc[vr]);
          sRH[m][col] = (__bf16)(s * (float)sXH[m][DD + col]);
        }
      } else {                                 // u gate -> f32 for blending
        const int col = (ntile - 16) * 16 + n16;
#pragma unroll
        for (int vr = 0; vr < 8; ++vr) {
          const int m = vr + 8 * g;
          sU[m][col] = sigmoidf_(acc[vr]);
        }
      }
    }
    __syncthreads();

    // ---- candidate GEMM: K = {x part from sXH | r*h part from sRH} ----
#pragma unroll
    for (int ti = 0; ti < 2; ++ti) {
      const int n = (wave + ti * 8) * 16 + n16;
      const float bias = cbias[ti];
      v8f acc = {bias, bias, bias, bias, bias, bias, bias, bias};
      const __bf16* wrow = cwrow[ti];
#pragma unroll
      for (int kt = 0; kt < KK / 32; ++kt) {
        v8bf a0, a1;
        if (kt < DD / 32) {                    // uniform branch
          const int ka = kt * 32 + g * 8;
          a0 = *(const v8bf*)&sXH[n16][ka];
          a1 = *(const v8bf*)&sXH[n16][ka + 16];
        } else {
          const int ka = (kt - DD / 32) * 32 + g * 8;
          a0 = *(const v8bf*)&sRH[n16][ka];
          a1 = *(const v8bf*)&sRH[n16][ka + 16];
        }
        const int kb = kt * 32 + g * 8;
        v8bf b0 = *(const v8bf*)(wrow + kb);
        v8bf b1 = *(const v8bf*)(wrow + kb + 16);
        acc = __builtin_amdgcn_wmma_f32_16x16x32_bf16(
            false, cat8(a0, a1), false, cat8(b0, b1), (short)0, acc, false, false);
      }
      // ---- blend + mask + output; h stays in LDS ----
#pragma unroll
      for (int vr = 0; vr < 8; ++vr) {
        const int m = vr + 8 * g;
        const float c  = tanhf(acc[vr]);
        const float u  = sU[m][n];
        const float ho = sH[m][n];
        const float nh = u * ho + (1.0f - u) * c;
        const bool valid = t < sLen[m];
        sH[m][n] = valid ? nh : ho;
        out[(((size_t)(b0 + m) * TT) + t) * HH + n] = valid ? nh : 0.0f;
      }
    }
    __syncthreads();   // protect sXH/sRH/sU/sH before next step's staging
  }
}

// ---------------------------------------------------------------------------
extern "C" void kernel_launch(void* const* d_in, const int* in_sizes, int n_in,
                              void* d_out, int out_size, void* d_ws, size_t ws_size,
                              hipStream_t stream) {
  const float* x      = (const float*)d_in[0];
  const int*   seqlen = (const int*)  d_in[1];
  const float* gk     = (const float*)d_in[2];
  const float* gb     = (const float*)d_in[3];
  const float* ck     = (const float*)d_in[4];
  const float* cb     = (const float*)d_in[5];
  float* out = (float*)d_out;

  char* ws = (char*)d_ws;
  __bf16* WgT = (__bf16*)ws;                 // 512*512*2 = 512KB
  __bf16* WcT = (__bf16*)(ws + 512 * 1024);  // 256*512*2 = 256KB

  gru_prep<<<dim3(1536), dim3(256), 0, stream>>>(gk, ck, WgT, WcT);
  gru_persistent<<<dim3(BB / 16), dim3(256), 0, stream>>>(
      x, seqlen, gb, cb, WgT, WcT, out);
}
